// RoboGraphormerLayer_88424786690349
// MI455X (gfx1250) — matrix operand order repaired and checked by reading, not compiled, not run on hardware
//
#include <hip/hip_runtime.h>
#include <math.h>

// ---------------------------------------------------------------------------
// RoboGraphormer layer for MI455X (gfx1250), wave32, WMMA bf16 path.
// B=4, N=256, D=256, H=8, HD=32, DFF=1024.
// ---------------------------------------------------------------------------

typedef __attribute__((ext_vector_type(16))) __bf16 v16bf;
typedef __attribute__((ext_vector_type(8)))  float  v8f;
typedef __attribute__((ext_vector_type(4)))  float  f32x4;

__device__ __forceinline__ v8f wmma_bf16(v16bf a, v16bf b, v8f c) {
  // D = A(16x32 bf16) * B(32x16 bf16) + C(16x16 f32)
  return __builtin_amdgcn_wmma_f32_16x16x32_bf16(
      /*neg_a=*/false, a, /*neg_b=*/false, b,
      /*c_mod=*/(short)0, c, /*reuse_a=*/false, /*reuse_b=*/false);
}

// A-fragment (16x32, M x K) from row-major f32 [16 x lda], K offset k0.
// Lane L: M = L&15, half = L>>4; idx i -> K = half*16 + i.
template <bool NT>
__device__ __forceinline__ v16bf load_a_frag_f32(const float* A, int lda, int k0) {
  const int lane = threadIdx.x & 31;
  const int m    = lane & 15;
  const int hf   = lane >> 4;
  const float* p = A + (size_t)m * lda + k0 + hf * 16;
  f32x4 a0, a1, a2, a3;
  if (NT) {
    a0 = __builtin_nontemporal_load((const f32x4*)(p));
    a1 = __builtin_nontemporal_load((const f32x4*)(p + 4));
    a2 = __builtin_nontemporal_load((const f32x4*)(p + 8));
    a3 = __builtin_nontemporal_load((const f32x4*)(p + 12));
  } else {
    a0 = *(const f32x4*)(p);
    a1 = *(const f32x4*)(p + 4);
    a2 = *(const f32x4*)(p + 8);
    a3 = *(const f32x4*)(p + 12);
  }
  v16bf f;
  f[0]  = (__bf16)a0[0]; f[1]  = (__bf16)a0[1]; f[2]  = (__bf16)a0[2]; f[3]  = (__bf16)a0[3];
  f[4]  = (__bf16)a1[0]; f[5]  = (__bf16)a1[1]; f[6]  = (__bf16)a1[2]; f[7]  = (__bf16)a1[3];
  f[8]  = (__bf16)a2[0]; f[9]  = (__bf16)a2[1]; f[10] = (__bf16)a2[2]; f[11] = (__bf16)a2[3];
  f[12] = (__bf16)a3[0]; f[13] = (__bf16)a3[1]; f[14] = (__bf16)a3[2]; f[15] = (__bf16)a3[3];
  return f;
}

// B-fragment (32x16, K x N) where element (k,n) lives at Bm[n*ldb + k] (f32,
// contiguous in k). Lane L: n = L&15, K base = (L>=16)?16:0, 16 contiguous k.
__device__ __forceinline__ v16bf load_b_frag_f32c(const float* Bm, int ldb) {
  const int lane = threadIdx.x & 31;
  const int n    = lane & 15;
  const int kb   = (lane >> 4) * 16;
  const float* p = Bm + (size_t)n * ldb + kb;
  f32x4 a0 = *(const f32x4*)(p);
  f32x4 a1 = *(const f32x4*)(p + 4);
  f32x4 a2 = *(const f32x4*)(p + 8);
  f32x4 a3 = *(const f32x4*)(p + 12);
  v16bf f;
  f[0]  = (__bf16)a0[0]; f[1]  = (__bf16)a0[1]; f[2]  = (__bf16)a0[2]; f[3]  = (__bf16)a0[3];
  f[4]  = (__bf16)a1[0]; f[5]  = (__bf16)a1[1]; f[6]  = (__bf16)a1[2]; f[7]  = (__bf16)a1[3];
  f[8]  = (__bf16)a2[0]; f[9]  = (__bf16)a2[1]; f[10] = (__bf16)a2[2]; f[11] = (__bf16)a2[3];
  f[12] = (__bf16)a3[0]; f[13] = (__bf16)a3[1]; f[14] = (__bf16)a3[2]; f[15] = (__bf16)a3[3];
  return f;
}

// ---------------------------------------------------------------------------
// Weight pre-pack: f32 [K x srcN] row-major -> bf16 WMMA-B fragment order.
// Tile (kt,nt) of 32x16 stored at ((nt*(K/32)+kt)*512); within tile index =
// lane*16 + i with lane = (kk>=16)*16 + nn, i = kk&15.  N is padded (>= srcN).
// ---------------------------------------------------------------------------
__global__ void k_pack(const float* __restrict__ src, __bf16* __restrict__ dst,
                       int K, int N, int srcN) {
  int idx = blockIdx.x * 256 + threadIdx.x;
  if (idx >= K * N) return;
  int k = idx / N, n = idx % N;
  float v = (n < srcN) ? src[(size_t)k * srcN + n] : 0.0f;
  int kt = k >> 5, kk = k & 31, nt = n >> 4, nn = n & 15;
  int lane = ((kk >> 4) << 4) + nn;
  int ii   = kk & 15;
  dst[((size_t)(nt * (K >> 5) + kt) << 9) + lane * 16 + ii] = (__bf16)v;
}

// ---------------------------------------------------------------------------
// Generic WMMA GEMM: out[R x N] = act(A[R x K] @ Bpacked + bias (+ res)).
// grid = (R/16, N/128), block = 256 (8 waves, 1 n-tile each).
// act: 0 = none, 2 = exact GELU.
// ---------------------------------------------------------------------------
__global__ void __launch_bounds__(256) k_gemm(
    const float* __restrict__ A, const __bf16* __restrict__ Bp,
    const float* __restrict__ bias, const float* __restrict__ res,
    float* __restrict__ out, int K, int N, int act) {
  const int wave = threadIdx.x >> 5;
  const int lane = threadIdx.x & 31;
  const int hf   = lane >> 4;
  const int nn   = lane & 15;
  const int r0   = blockIdx.x << 4;
  const int nt   = blockIdx.y * 8 + wave;
  const int ktiles = K >> 5;
  const float* Arow = A + (size_t)r0 * K;
  const v16bf* Bv = (const v16bf*)Bp;

  v8f acc = {};
  for (int kt = 0; kt < ktiles; ++kt) {
    v16bf a = load_a_frag_f32<false>(Arow, K, kt * 32);
    v16bf b = Bv[((size_t)nt * ktiles + kt) * 32 + lane];
    acc = wmma_bf16(a, b, acc);
  }
  const int n = nt * 16 + nn;
  const float bv = bias[n];
#pragma unroll
  for (int r = 0; r < 8; ++r) {
    int m = r + hf * 8;
    float v = acc[r] + bv;
    if (res) v += res[(size_t)(r0 + m) * N + n];
    if (act == 2) v = 0.5f * v * (1.0f + erff(v * 0.70710678118654752f));
    out[(size_t)(r0 + m) * N + n] = v;
  }
}

// ---------------------------------------------------------------------------
// LayerNorm over D=256.  grid = R rows, block = 32 (one wave per row).
// ---------------------------------------------------------------------------
__global__ void __launch_bounds__(32) k_ln(const float* __restrict__ x,
                                           const float* __restrict__ g,
                                           const float* __restrict__ bta,
                                           float* __restrict__ y) {
  const int row  = blockIdx.x;
  const int lane = threadIdx.x;
  const float* p = x + (size_t)row * 256 + lane * 8;
  f32x4 v0 = *(const f32x4*)(p);
  f32x4 v1 = *(const f32x4*)(p + 4);
  float vals[8] = {v0[0], v0[1], v0[2], v0[3], v1[0], v1[1], v1[2], v1[3]};
  float s = 0.f;
#pragma unroll
  for (int t = 0; t < 8; ++t) s += vals[t];
  for (int mask = 1; mask < 32; mask <<= 1) s += __shfl_xor(s, mask, 32);
  const float mean = s * (1.0f / 256.0f);
  float vs = 0.f;
#pragma unroll
  for (int t = 0; t < 8; ++t) { float d = vals[t] - mean; vs += d * d; }
  for (int mask = 1; mask < 32; mask <<= 1) vs += __shfl_xor(vs, mask, 32);
  const float rs = rsqrtf(vs * (1.0f / 256.0f) + 1e-5f);
#pragma unroll
  for (int t = 0; t < 8; ++t) {
    int n = lane * 8 + t;
    y[(size_t)row * 256 + n] = (vals[t] - mean) * rs * g[n] + bta[n];
  }
}

// ---------------------------------------------------------------------------
// V transpose: qkv[., 512 + h*32+hd] -> vT[b,h,hd,j]  (for contiguous-k B frags)
// ---------------------------------------------------------------------------
__global__ void __launch_bounds__(256) k_vt(const float* __restrict__ qkv,
                                            float* __restrict__ vT) {
  int idx = blockIdx.x * 256 + threadIdx.x;   // 0 .. 262143
  int c   = idx & 255;
  int row = idx >> 8;                         // b*256 + j
  int b = row >> 8, j = row & 255;
  int h = c >> 5, hd = c & 31;
  vT[((size_t)((b * 8 + h) * 32 + hd) << 8) + j] =
      qkv[(size_t)row * 768 + 512 + c];
}

// ---------------------------------------------------------------------------
// Edge kernel (dominant).  One wave owns one 16-row stripe (fixed b,i; 16 j's)
// and ALL 16 n-tiles (16 accumulators): each A fragment is loaded once (NT)
// and reused 16x from registers -> 8x less L0 traffic than wave-sharing.
//   e_new = relu(e @ We + be + hs[i] + hd[j])  (NT stores, streamed once)
//   fused: bias[b, :, i, j-stripe] = e_new @ Wb(pad16) + bb  via per-wave
//   LDS restaging chunks of 16x32 (DS ops are in-order within a wave).
// block = 256 (8 waves = 8 M-tiles), grid = B*N*N/16/8 = 2048.
// ---------------------------------------------------------------------------
#define PS_LD 36   // 16x36 f32 per-wave staging; conflict-free, 16B-aligned rows
__global__ void __launch_bounds__(256) k_edge(
    const float* __restrict__ e, const __bf16* __restrict__ pWe,
    const __bf16* __restrict__ pWb, const float* __restrict__ be,
    const float* __restrict__ hsrc, const float* __restrict__ hdst,
    const float* __restrict__ bb, float* __restrict__ e_out,
    float* __restrict__ biasT) {
  __shared__ float ps[8][16 * PS_LD];
  const int wave = threadIdx.x >> 5;
  const int lane = threadIdx.x & 31;
  const int hf   = lane >> 4;
  const int nn   = lane & 15;
  const int mt   = blockIdx.x * 8 + wave;
  const int r0   = mt << 4;
  const int b    = r0 >> 16;
  const int i    = (r0 >> 8) & 255;
  const int j0   = r0 & 255;
  const float* Arow = e + (size_t)r0 * 256;
  const v16bf* Wv  = (const v16bf*)pWe;
  const v16bf* Wbv = (const v16bf*)pWb;
  float* myps = &ps[wave][0];

  v8f acc[16];
  v8f zero = {};
#pragma unroll
  for (int t = 0; t < 16; ++t) acc[t] = zero;

  for (int kt = 0; kt < 8; ++kt) {
    v16bf a = load_a_frag_f32<true>(Arow, 256, kt * 32);   // NT: stream e once
#pragma unroll
    for (int nt = 0; nt < 16; ++nt)
      acc[nt] = wmma_bf16(a, Wv[(nt * 8 + kt) * 32 + lane], acc[nt]);
  }

  const float* hsrow = hsrc + (size_t)(b * 256 + i) * 256;
  v8f bacc = zero;
#pragma unroll
  for (int c = 0; c < 8; ++c) {        // process n in chunks of 32 (2 n-tiles)
#pragma unroll
    for (int t = 0; t < 2; ++t) {
      const int nt = c * 2 + t;
      const int n  = nt * 16 + nn;
      const float base = be[n] + hsrow[n];
      v8f a = acc[nt];
#pragma unroll
      for (int r = 0; r < 8; ++r) {
        int m = r + hf * 8;
        float v = a[r] + base + hdst[(size_t)(b * 256 + j0 + m) * 256 + n];
        v = fmaxf(v, 0.0f);
        __builtin_nontemporal_store(v, &e_out[(size_t)(r0 + m) * 256 + n]);
        myps[m * PS_LD + t * 16 + nn] = v;
      }
    }
    // bias partial: (16 x 32 chunk of e_new) @ (32 x 16 chunk of Wb-padded)
    v16bf af = load_a_frag_f32<false>(myps, PS_LD, 0);
    bacc = wmma_bf16(af, Wbv[c * 32 + lane], bacc);
  }
  if (nn < 8) {
    const float bbv = bb[nn];
#pragma unroll
    for (int r = 0; r < 8; ++r) {
      int m = r + hf * 8;
      biasT[((size_t)(b * 8 + nn) * 256 + i) * 256 + (j0 + m)] = bacc[r] + bbv;
    }
  }
}

// ---------------------------------------------------------------------------
// Flash-style attention with additive bias.  One wave per (b, h, i-tile of 16).
// K-dim HD=32 -> one WMMA per QK^T half; P(16x32)@V via LDS relayout + WMMA.
// grid = B*H*(N/16) = 512, block = 32.
// ---------------------------------------------------------------------------
__global__ void __launch_bounds__(32) k_attn(const float* __restrict__ qkv,
                                             const float* __restrict__ vT,
                                             const float* __restrict__ biasT,
                                             float* __restrict__ o) {
  __shared__ float ps[16 * PS_LD];
  const int bid = blockIdx.x;
  const int it = bid & 15, h = (bid >> 4) & 7, b = bid >> 7;
  const int i0 = it << 4;
  const int lane = threadIdx.x;
  const int hf = lane >> 4, nn = lane & 15;

  v16bf qa = load_a_frag_f32<false>(qkv + (size_t)(b * 256 + i0) * 768 + h * 32, 768, 0);

  float mrow[8], lrow[8];
#pragma unroll
  for (int r = 0; r < 8; ++r) { mrow[r] = -3.0e38f; lrow[r] = 0.0f; }
  v8f o0 = {}, o1 = {};
  const float* biasB = biasT + (size_t)(b * 8 + h) * 256 * 256;
  const float scale = 0.17677669529663687f;   // 1/sqrt(32)

  for (int jt = 0; jt < 8; ++jt) {
    const int j0 = jt * 32;
    v16bf kb0 = load_b_frag_f32c(qkv + (size_t)(b * 256 + j0) * 768 + 256 + h * 32, 768);
    v16bf kb1 = load_b_frag_f32c(qkv + (size_t)(b * 256 + j0 + 16) * 768 + 256 + h * 32, 768);
    v8f z = {};
    v8f s0 = wmma_bf16(qa, kb0, z);
    v8f s1 = wmma_bf16(qa, kb1, z);

    float p0[8], p1[8];
#pragma unroll
    for (int r = 0; r < 8; ++r) {
      int m = r + hf * 8;
      p0[r] = s0[r] * scale + biasB[(size_t)(i0 + m) * 256 + j0 + nn];
      p1[r] = s1[r] * scale + biasB[(size_t)(i0 + m) * 256 + j0 + 16 + nn];
    }
#pragma unroll
    for (int r = 0; r < 8; ++r) {
      float cm = fmaxf(p0[r], p1[r]);
      cm = fmaxf(cm, __shfl_xor(cm, 1, 32));
      cm = fmaxf(cm, __shfl_xor(cm, 2, 32));
      cm = fmaxf(cm, __shfl_xor(cm, 4, 32));
      cm = fmaxf(cm, __shfl_xor(cm, 8, 32));
      float nm = fmaxf(mrow[r], cm);
      float alpha = __expf(mrow[r] - nm);
      mrow[r] = nm;
      p0[r] = __expf(p0[r] - nm);
      p1[r] = __expf(p1[r] - nm);
      float rs = p0[r] + p1[r];
      rs += __shfl_xor(rs, 1, 32);
      rs += __shfl_xor(rs, 2, 32);
      rs += __shfl_xor(rs, 4, 32);
      rs += __shfl_xor(rs, 8, 32);
      lrow[r] = lrow[r] * alpha + rs;
      o0[r] *= alpha;
      o1[r] *= alpha;
    }
#pragma unroll
    for (int r = 0; r < 8; ++r) {       // C-layout -> LDS (A-layout source)
      int m = r + hf * 8;
      ps[m * PS_LD + nn]      = p0[r];
      ps[m * PS_LD + 16 + nn] = p1[r];
    }
    __syncthreads();
    v16bf pa  = load_a_frag_f32<false>(ps, PS_LD, 0);
    v16bf vb0 = load_b_frag_f32c(vT + (size_t)((b * 8 + h) * 32) * 256 + j0, 256);
    v16bf vb1 = load_b_frag_f32c(vT + (size_t)((b * 8 + h) * 32 + 16) * 256 + j0, 256);
    o0 = wmma_bf16(pa, vb0, o0);
    o1 = wmma_bf16(pa, vb1, o1);
    __syncthreads();
  }
#pragma unroll
  for (int r = 0; r < 8; ++r) {
    int m = r + hf * 8;
    float inv = 1.0f / lrow[r];
    o[(size_t)(b * 256 + i0 + m) * 256 + h * 32 + nn]      = o0[r] * inv;
    o[(size_t)(b * 256 + i0 + m) * 256 + h * 32 + 16 + nn] = o1[r] * inv;
  }
}

// ---------------------------------------------------------------------------
// Host-side orchestration
// ---------------------------------------------------------------------------
extern "C" void kernel_launch(void* const* d_in, const int* in_sizes, int n_in,
                              void* d_out, int out_size, void* d_ws, size_t ws_size,
                              hipStream_t stream) {
  const float* h    = (const float*)d_in[0];
  const float* e    = (const float*)d_in[1];
  const float* We   = (const float*)d_in[2];
  const float* be   = (const float*)d_in[3];
  const float* Ws   = (const float*)d_in[4];
  const float* bs   = (const float*)d_in[5];
  const float* Wd   = (const float*)d_in[6];
  const float* bd   = (const float*)d_in[7];
  const float* Wb   = (const float*)d_in[8];
  const float* bb   = (const float*)d_in[9];
  const float* ln1g = (const float*)d_in[10];
  const float* ln1b = (const float*)d_in[11];
  const float* Wqkv = (const float*)d_in[12];
  const float* bqkv = (const float*)d_in[13];
  const float* Wo   = (const float*)d_in[14];
  const float* bo   = (const float*)d_in[15];
  const float* ln2g = (const float*)d_in[16];
  const float* ln2b = (const float*)d_in[17];
  const float* W1   = (const float*)d_in[18];
  const float* b1   = (const float*)d_in[19];
  const float* W2   = (const float*)d_in[20];
  const float* b2   = (const float*)d_in[21];

  char* ws = (char*)d_ws;
  // bf16 packed weights
  __bf16* pWe   = (__bf16*)(ws + 0);
  __bf16* pWs   = (__bf16*)(ws + 131072);
  __bf16* pWd   = (__bf16*)(ws + 262144);
  __bf16* pWqkv = (__bf16*)(ws + 393216);
  __bf16* pWo   = (__bf16*)(ws + 786432);
  __bf16* pW1   = (__bf16*)(ws + 917504);
  __bf16* pW2   = (__bf16*)(ws + 1441792);
  __bf16* pWb   = (__bf16*)(ws + 1966080);
  // f32 activations
  float* hs_b  = (float*)(ws + 1974272);   // [1024,256]
  float* hd_b  = (float*)(ws + 3022848);   // [1024,256]
  float* x_b   = (float*)(ws + 4071424);   // ln1 out
  float* qkv_b = (float*)(ws + 5120000);   // [1024,768]
  float* vT_b  = (float*)(ws + 8265728);   // [B,H,HD,N]
  float* bias_b= (float*)(ws + 9314304);   // [B,H,N,N]
  float* o_b   = (float*)(ws + 17702912);  // [1024,256]
  float* h1_b  = (float*)(ws + 18751488);
  float* y_b   = (float*)(ws + 19800064);
  float* t_b   = (float*)(ws + 20848640);  // [1024,1024]

  float* h2_out = (float*)d_out;                 // [4,256,256]
  float* e_out  = (float*)d_out + 262144;        // [4,256,256,256]

  // --- pack weights to bf16 fragment order ---
  k_pack<<<256,  256, 0, stream>>>(We,   pWe,   256, 256,  256);
  k_pack<<<256,  256, 0, stream>>>(Ws,   pWs,   256, 256,  256);
  k_pack<<<256,  256, 0, stream>>>(Wd,   pWd,   256, 256,  256);
  k_pack<<<768,  256, 0, stream>>>(Wqkv, pWqkv, 256, 768,  768);
  k_pack<<<256,  256, 0, stream>>>(Wo,   pWo,   256, 256,  256);
  k_pack<<<1024, 256, 0, stream>>>(W1,   pW1,   256, 1024, 1024);
  k_pack<<<1024, 256, 0, stream>>>(W2,   pW2,   1024, 256, 256);
  k_pack<<<16,   256, 0, stream>>>(Wb,   pWb,   256, 16,   8);     // pad H=8 -> 16

  // --- node projections hs = h@Ws+bs, hd = h@Wd+bd ---
  k_gemm<<<dim3(64, 2), 256, 0, stream>>>(h, pWs, bs, nullptr, hs_b, 256, 256, 0);
  k_gemm<<<dim3(64, 2), 256, 0, stream>>>(h, pWd, bd, nullptr, hd_b, 256, 256, 0);

  // --- attention input path: x = LN1(h); qkv = x@Wqkv+bqkv; vT ---
  k_ln<<<1024, 32, 0, stream>>>(h, ln1g, ln1b, x_b);
  k_gemm<<<dim3(64, 6), 256, 0, stream>>>(x_b, pWqkv, bqkv, nullptr, qkv_b, 256, 768, 0);
  k_vt<<<1024, 256, 0, stream>>>(qkv_b, vT_b);

  // --- dominant edge GEMM + fused per-head bias projection ---
  k_edge<<<2048, 256, 0, stream>>>(e, pWe, pWb, be, hs_b, hd_b, bb, e_out, bias_b);

  // --- attention + output proj + residual ---
  k_attn<<<512, 32, 0, stream>>>(qkv_b, vT_b, bias_b, o_b);
  k_gemm<<<dim3(64, 2), 256, 0, stream>>>(o_b, pWo, bo, h, h1_b, 256, 256, 0);

  // --- FFN: y = LN2(h1); t = gelu(y@W1+b1); h2 = h1 + t@W2+b2 ---
  k_ln<<<1024, 32, 0, stream>>>(h1_b, ln2g, ln2b, y_b);
  k_gemm<<<dim3(64, 8), 256, 0, stream>>>(y_b, pW1, b1, nullptr, t_b, 256, 1024, 2);
  k_gemm<<<dim3(64, 2), 256, 0, stream>>>(t_b, pW2, b2, h1_b, h2_out, 1024, 256, 0);
}
